// TransformerEncoderLayer_52673478918516
// MI455X (gfx1250) — matrix operand order repaired and checked by reading, not compiled, not run on hardware
//
#include <hip/hip_runtime.h>
#include <hip/hip_bf16.h>

#define B_   2
#define S_   2048
#define D_   1024
#define H_   16
#define DEPTH_ 64
#define FF_  4096
#define M_   (B_ * S_)   // 4096 rows

typedef _Float16 f16;
typedef __attribute__((ext_vector_type(16))) _Float16 v16h;
typedef __attribute__((ext_vector_type(8)))  _Float16 v8h;
typedef __attribute__((ext_vector_type(8)))  float    v8f;
typedef __attribute__((ext_vector_type(4)))  unsigned int v4u;
typedef __attribute__((ext_vector_type(8)))  unsigned int v8u;

__device__ __forceinline__ v8f wmma_f32_16x16x32(v16h a, v16h b, v8f c) {
  return __builtin_amdgcn_wmma_f32_16x16x32_f16(
      /*neg_a=*/false, a, /*neg_b=*/false, b,
      /*c_mod=*/(short)0, c, /*reuse_a=*/false, /*reuse_b=*/false);
}

// ---------------------------------------------------------------------------
// Tensor Data Mover: one instruction DMAs an 8KB contiguous block into LDS.
// D# built per CDNA5 ISA (group0 bits: count=1, lds_addr,
// 57b global_addr, type=2; group1: data_size=1 (2B), tensor 4096x1,
// tile 4096x1, dim0 stride 4096). Issued wave-uniform (SGPR groups).
// ---------------------------------------------------------------------------
__device__ __forceinline__ void tdm_load_8kb(const void* gsrc, void* ldst) {
  uint64_t ga = (uint64_t)(uintptr_t)gsrc;
  uint32_t la = (uint32_t)(uintptr_t)ldst;   // low 32 bits of generic = LDS byte addr
  v4u g0;
  g0[0] = 1u;                                  // count=1 (valid user descriptor)
  g0[1] = la;                                  // lds_addr
  g0[2] = (uint32_t)ga;                        // global_addr[31:0]
  g0[3] = (uint32_t)(ga >> 32) | (2u << 30);   // global_addr[56:32] | type=2
  v8u g1;
  g1[0] = (1u << 16);        // workgroup_mask=0, data_size=1 (2 bytes)
  g1[1] = (4096u << 16);     // atomic_barrier_addr=0 | tensor_dim0[15:0]=4096
  g1[2] = (1u << 16);        // tensor_dim0[31:16]=0 | tensor_dim1[15:0]=1
  g1[3] = (4096u << 16);     // tensor_dim1[31:16]=0 | tile_dim0=4096
  g1[4] = 1u;                // tile_dim1=1 | tile_dim2=0
  g1[5] = 4096u;             // tensor_dim0_stride[31:0]=4096
  g1[6] = 0u;                // stride0[47:32]=0 | tensor_dim1_stride[15:0]=0
  g1[7] = 0u;
  v4u z = {0u, 0u, 0u, 0u};  // groups 2/3: dims/tiles zero (unused for 2D tile)
  asm volatile("tensor_load_to_lds %0, %1, %2, %3"
               :: "s"(g0), "s"(g1), "s"(z), "s"(z)
               : "memory");
}
__device__ __forceinline__ void wait_tensorcnt0() {
  __builtin_amdgcn_s_wait_tensorcnt(0);
}

// Butterfly max over the 16-lane half using DPP (no LDS, no waits).
__device__ __forceinline__ float dpp_max16(float x) {
  int t;
  t = __builtin_amdgcn_update_dpp(0, __float_as_int(x), 0xB1, 0xF, 0xF, true);
  x = fmaxf(x, __int_as_float(t));
  t = __builtin_amdgcn_update_dpp(0, __float_as_int(x), 0x4E, 0xF, 0xF, true);
  x = fmaxf(x, __int_as_float(t));
  t = __builtin_amdgcn_update_dpp(0, __float_as_int(x), 0x141, 0xF, 0xF, true);
  x = fmaxf(x, __int_as_float(t));
  t = __builtin_amdgcn_update_dpp(0, __float_as_int(x), 0x140, 0xF, 0xF, true);
  x = fmaxf(x, __int_as_float(t));
  return x;
}

// ---------------------------------------------------------------------------
__global__ void cast_f32_f16_k(const float* __restrict__ in, f16* __restrict__ out, int n) {
  int i = blockIdx.x * blockDim.x + threadIdx.x;
  if (i < n) out[i] = (f16)in[i];
}

// ---------------------------------------------------------------------------
// Pack fp32 weight W[K][N] (row-major) into WMMA B-fragment layout (f16):
// packed[((kb*NT + nt)*32 + lane)*16 + e] = W[kb*32 + (e<8 ? 0 : 16) + (lane>=16?8:0) + (e&7)][nt*16 + lane%16]
// ---------------------------------------------------------------------------
__global__ void pack_w_k(const float* __restrict__ W, f16* __restrict__ P, int K, int N) {
  int idx = blockIdx.x * blockDim.x + threadIdx.x;
  if (idx >= K * N) return;
  int e    = idx & 15;
  int lane = (idx >> 4) & 31;
  int t    = idx >> 9;              // tile id = kb*NT + nt
  int NT   = N >> 4;
  int kb   = t / NT, nt = t % NT;
  int k    = kb * 32 + ((e < 8) ? 0 : 16) + ((lane >= 16) ? 8 : 0) + (e & 7);
  int n    = nt * 16 + (lane & 15);
  P[idx] = (f16)W[(size_t)k * N + n];
}

// ---------------------------------------------------------------------------
// GEMM: out = A(f16, MxK row-major) @ Wpacked + bias.
// Block = 4 waves, 64(M) x 128(N) tile. The 8 packed B tiles per k-step form
// one contiguous 8KB block, DMA'd into LDS by the Tensor Data Mover (wave 0
// issues; TENSORcnt + barrier handoff), double-buffered and prefetched one
// k-step ahead. 8 WMMAs per wave per k-step amortize the barrier.
// grid = (N/128, M/64), blockDim = 128
// ---------------------------------------------------------------------------
template <bool RELU, bool OUT32, bool OUT16>
__global__ void gemm_wmma_k(const f16* __restrict__ A, const f16* __restrict__ Wp,
                            const float* __restrict__ bias,
                            float* __restrict__ out32, f16* __restrict__ out16,
                            int Mdim, int Kdim, int Ndim) {
  const int tid    = threadIdx.x;
  const int lane   = tid & 31;
  const int wave   = tid >> 5;
  const int mtile  = blockIdx.y * 4 + wave;
  const int nt0    = blockIdx.x * 8;
  const int NT     = Ndim >> 4;
  const int KB     = Kdim >> 5;
  const int half16 = lane >> 4;
  const int ln     = lane & 15;
  const bool issuer = (wave == 0);

  __shared__ __align__(256) f16 bstage[2][4096];   // 2 x 8KB double buffer

  auto prefetch = [&](int kb, int buf) {
    if (issuer)
      tdm_load_8kb(Wp + ((size_t)kb * NT + nt0) * 512, &bstage[buf][0]);
  };

  v8f acc[8] = {};
  const int arow = mtile * 16 + ln;
  const f16* arowp = A + (size_t)arow * Kdim + half16 * 8;

  prefetch(0, 0);
  if (issuer) wait_tensorcnt0();
  __syncthreads();

  for (int kb = 0; kb < KB; ++kb) {
    const int cur = kb & 1;
    if (kb + 1 < KB) prefetch(kb + 1, cur ^ 1);

    // A fragment: lane<16 holds row ln, K {0..7, 16..23}; lane>=16 holds K {8..15, 24..31}
    v8h alo = *(const v8h*)(arowp + kb * 32);
    v8h ahi = *(const v8h*)(arowp + kb * 32 + 16);
    v16h a;
#pragma unroll
    for (int i = 0; i < 8; ++i) { a[i] = alo[i]; a[i + 8] = ahi[i]; }

    const f16* bb = &bstage[cur][0];
#pragma unroll
    for (int j = 0; j < 8; ++j) {
      v16h b = *(const v16h*)(bb + (j * 32 + lane) * 16);
      acc[j] = wmma_f32_16x16x32(a, b, acc[j]);
    }

    if (kb + 1 < KB) {
      if (issuer) wait_tensorcnt0();  // next buffer fully landed
      __syncthreads();                // publish; readers of cur are done
    }
  }

  // C layout: VGPR r, lanes0-15 -> row r, lanes16-31 -> row r+8; col = lane%16
#pragma unroll
  for (int j = 0; j < 8; ++j) {
    int col  = (nt0 + j) * 16 + ln;
    float bv = bias[col];
#pragma unroll
    for (int r = 0; r < 8; ++r) {
      int row = mtile * 16 + half16 * 8 + r;
      float v = acc[j][r] + bv;
      if (RELU) v = v > 0.f ? v : 0.f;
      if (OUT32) out32[(size_t)row * Ndim + col] = v;
      if (OUT16) out16[(size_t)row * Ndim + col] = (f16)v;
    }
  }
}

// ---------------------------------------------------------------------------
// Transpose V (f16 [B,S,H*64]) -> vT (f16 [B*H, 64, S]) for contiguous PV B-fragments
// ---------------------------------------------------------------------------
__global__ void transpose_v_k(const f16* __restrict__ v, f16* __restrict__ vT) {
  int i = blockIdx.x * blockDim.x + threadIdx.x;
  if (i >= B_ * S_ * D_) return;
  int d = i & 63;
  int h = (i >> 6) & 15;
  int s = (i >> 10) & (S_ - 1);
  int b = i >> 21;
  vT[(((size_t)(b * H_ + h) * DEPTH_) + d) * S_ + s] = v[i];
}

// ---------------------------------------------------------------------------
// Fused flash attention, 64-key blocks. K fragments loaded up front; V
// fragment loads issued before the softmax so their latency hides under the
// exp/max VALU chain. Row-max via DPP butterflies; row-sum via WMMA against
// an all-ones B fragment. grid = (S/64, B*H), blockDim = 128 (4 waves).
// ---------------------------------------------------------------------------
__global__ void attention_k(const f16* __restrict__ q, const f16* __restrict__ k,
                            const f16* __restrict__ vT, f16* __restrict__ ctx) {
  const int lane   = threadIdx.x & 31;
  const int wave   = threadIdx.x >> 5;
  const int bh     = blockIdx.y;
  const int b      = bh >> 4, h = bh & 15;
  const int qtile  = blockIdx.x * 4 + wave;
  const int half16 = lane >> 4;
  const int ln     = lane & 15;

  __shared__ __align__(32) f16 lds_p[4][16 * 64];   // per-wave P staging (16 rows x 64 keys)
  f16* lp = lds_p[wave];

  // Q A-fragments (depth 0..31 and 32..63), pre-scaled by 1/sqrt(64) = 0.125
  const int qrow = qtile * 16 + ln;
  const f16* qp = q + ((size_t)(b * S_ + qrow)) * D_ + h * DEPTH_ + half16 * 16;
  v16h a0, a1;
  {
    v8h l0 = *(const v8h*)(qp);
    v8h h0 = *(const v8h*)(qp + 8);
    v8h l1 = *(const v8h*)(qp + 32);
    v8h h1 = *(const v8h*)(qp + 40);
    const f16 sc = (f16)0.125f;
#pragma unroll
    for (int i = 0; i < 8; ++i) {
      a0[i] = l0[i] * sc; a0[i + 8] = h0[i] * sc;
      a1[i] = l1[i] * sc; a1[i + 8] = h1[i] * sc;
    }
  }

  v16h ones;
#pragma unroll
  for (int i = 0; i < 16; ++i) ones[i] = (f16)1.0f;

  float m[8], l[8];
  v8f acc[4] = {};
#pragma unroll
  for (int r = 0; r < 8; ++r) { m[r] = -1e30f; l[r] = 0.f; }

  const f16* kbasep = k + (size_t)b * S_ * D_ + h * DEPTH_;
  const f16* vTbase = vT + (size_t)bh * DEPTH_ * S_;

  for (int kb = 0; kb < S_ / 64; ++kb) {
    const int key0 = kb * 64;

    // ---- load all 8 K fragments, then run the 8 score WMMAs back-to-back
    v16h kf0[4], kf1[4];
#pragma unroll
    for (int t = 0; t < 4; ++t) {
      const f16* kp = kbasep + (size_t)(key0 + t * 16 + ln) * D_ + half16 * 16;
      kf0[t] = *(const v16h*)(kp);
      kf1[t] = *(const v16h*)(kp + 32);
    }
    v8f sv[4];
#pragma unroll
    for (int t = 0; t < 4; ++t) {
      v8f s = {};
      s = wmma_f32_16x16x32(a0, kf0[t], s);
      s = wmma_f32_16x16x32(a1, kf1[t], s);
      sv[t] = s;
    }

    // ---- kick off V fragment loads now; latency hides under the softmax
    v16h vf0[4], vf1[4];
#pragma unroll
    for (int j = 0; j < 4; ++j) {
      const f16* vp = vTbase + (size_t)(j * 16 + ln) * S_ + key0 + half16 * 16;
      vf0[j] = *(const v16h*)(vp);
      vf1[j] = *(const v16h*)(vp + 32);
    }

    // ---- online softmax: rows live per-lane (r + half16*8)
#pragma unroll
    for (int r = 0; r < 8; ++r) {
      float mx = fmaxf(fmaxf(sv[0][r], sv[1][r]), fmaxf(sv[2][r], sv[3][r]));
      mx = dpp_max16(mx);
      float mnew = fmaxf(m[r], mx);
      float corr = __expf(m[r] - mnew);
      m[r] = mnew;
      l[r] *= corr;
#pragma unroll
      for (int j = 0; j < 4; ++j) acc[j][r] *= corr;
      int row = r + half16 * 8;
#pragma unroll
      for (int t = 0; t < 4; ++t) {
        float p = __expf(sv[t][r] - mnew);
        lp[row * 64 + t * 16 + ln] = (f16)p;
      }
    }

    // ---- read P back as two A-fragments (keys 0..31 and 32..63)
    v16h pa0, pa1;
    {
      const f16* pr = lp + ln * 64 + half16 * 16;
      v8h lo0 = *(const v8h*)(pr);
      v8h hi0 = *(const v8h*)(pr + 8);
      v8h lo1 = *(const v8h*)(pr + 32);
      v8h hi1 = *(const v8h*)(pr + 40);
#pragma unroll
      for (int i = 0; i < 8; ++i) {
        pa0[i] = lo0[i]; pa0[i + 8] = hi0[i];
        pa1[i] = lo1[i]; pa1[i + 8] = hi1[i];
      }
    }

    // ---- row-sum of P via WMMA against all-ones B (C layout == state layout)
    v8f rsv = {};
    rsv = wmma_f32_16x16x32(pa0, ones, rsv);
    rsv = wmma_f32_16x16x32(pa1, ones, rsv);
#pragma unroll
    for (int r = 0; r < 8; ++r) l[r] += rsv[r];

    // ---- ctx += P(16x64) @ V(64x64)
#pragma unroll
    for (int j = 0; j < 4; ++j) {
      acc[j] = wmma_f32_16x16x32(pa0, vf0[j], acc[j]);
      acc[j] = wmma_f32_16x16x32(pa1, vf1[j], acc[j]);
    }
  }

  // normalize and write ctx (f16) back into [B,S,D] layout
  f16* cbase = ctx + ((size_t)(b * S_ + qtile * 16)) * D_ + h * DEPTH_;
#pragma unroll
  for (int r = 0; r < 8; ++r) {
    float inv = 1.f / l[r];
    int row = r + half16 * 8;
#pragma unroll
    for (int j = 0; j < 4; ++j)
      cbase[(size_t)row * D_ + j * 16 + ln] = (f16)(acc[j][r] * inv);
  }
}

// ---------------------------------------------------------------------------
// Residual + LayerNorm over D=1024; one 256-thread block per row.
// ---------------------------------------------------------------------------
__global__ void add_ln_k(const float* __restrict__ xres, const float* __restrict__ y,
                         const float* __restrict__ g, const float* __restrict__ be,
                         float* __restrict__ out32, f16* __restrict__ out16) {
  const int row = blockIdx.x;
  const int tid = threadIdx.x;
  __shared__ float red[8];
  const float* xr = xres + (size_t)row * D_;
  const float* yr = y + (size_t)row * D_;
  float vals[4];
  float s = 0.f;
#pragma unroll
  for (int i = 0; i < 4; ++i) {
    int c = tid + i * 256;
    float v = xr[c] + yr[c];
    vals[i] = v; s += v;
  }
  for (int off = 16; off > 0; off >>= 1) s += __shfl_xor(s, off, 32);
  if ((tid & 31) == 0) red[tid >> 5] = s;
  __syncthreads();
  float tot = 0.f;
#pragma unroll
  for (int w = 0; w < 8; ++w) tot += red[w];
  float mu = tot / (float)D_;
  __syncthreads();
  float vs = 0.f;
#pragma unroll
  for (int i = 0; i < 4; ++i) { float d0 = vals[i] - mu; vs += d0 * d0; }
  for (int off = 16; off > 0; off >>= 1) vs += __shfl_xor(vs, off, 32);
  if ((tid & 31) == 0) red[tid >> 5] = vs;
  __syncthreads();
  float vt = 0.f;
#pragma unroll
  for (int w = 0; w < 8; ++w) vt += red[w];
  float rstd = rsqrtf(vt / (float)D_ + 1e-5f);
#pragma unroll
  for (int i = 0; i < 4; ++i) {
    int c = tid + i * 256;
    float o = (vals[i] - mu) * rstd * g[c] + be[c];
    out32[(size_t)row * D_ + c] = o;
    if (out16) out16[(size_t)row * D_ + c] = (f16)o;
  }
}

// ---------------------------------------------------------------------------
extern "C" void kernel_launch(void* const* d_in, const int* in_sizes, int n_in,
                              void* d_out, int out_size, void* d_ws, size_t ws_size,
                              hipStream_t stream) {
  const float* x  = (const float*)d_in[0];
  const float* wq = (const float*)d_in[1];  const float* bq = (const float*)d_in[2];
  const float* wk = (const float*)d_in[3];  const float* bk = (const float*)d_in[4];
  const float* wv = (const float*)d_in[5];  const float* bv = (const float*)d_in[6];
  const float* wo = (const float*)d_in[7];  const float* bo = (const float*)d_in[8];
  const float* w1 = (const float*)d_in[9];  const float* b1 = (const float*)d_in[10];
  const float* w2 = (const float*)d_in[11]; const float* b2 = (const float*)d_in[12];
  const float* g1 = (const float*)d_in[13]; const float* be1 = (const float*)d_in[14];
  const float* g2 = (const float*)d_in[15]; const float* be2 = (const float*)d_in[16];
  float* out = (float*)d_out;

  char* wsb = (char*)d_ws;
  size_t off = 0;
  auto alloc = [&](size_t bytes) -> void* {
    void* p = wsb + off;
    off += (bytes + 255) & ~(size_t)255;
    return p;
  };
  const size_t MD = (size_t)M_ * D_;
  f16* xh    = (f16*)alloc(MD * 2);
  f16* wqp   = (f16*)alloc((size_t)D_ * D_ * 2);
  f16* wkp   = (f16*)alloc((size_t)D_ * D_ * 2);
  f16* wvp   = (f16*)alloc((size_t)D_ * D_ * 2);
  f16* wop   = (f16*)alloc((size_t)D_ * D_ * 2);
  f16* w1p   = (f16*)alloc((size_t)D_ * FF_ * 2);
  f16* w2p   = (f16*)alloc((size_t)FF_ * D_ * 2);
  f16* qh    = (f16*)alloc(MD * 2);
  f16* kh    = (f16*)alloc(MD * 2);
  f16* vh    = (f16*)alloc(MD * 2);
  f16* vT    = (f16*)alloc(MD * 2);
  f16* ctx   = (f16*)alloc(MD * 2);
  float* attn_out = (float*)alloc(MD * 4);
  float* out1     = (float*)alloc(MD * 4);
  f16*   out1h    = (f16*)alloc(MD * 2);
  f16*   ffh      = (f16*)alloc((size_t)M_ * FF_ * 2);
  float* ffn_out  = (float*)alloc(MD * 4);

  // 1) cast x, pack weights into WMMA B-fragment layout
  cast_f32_f16_k<<<(int)(MD / 256), 256, 0, stream>>>(x, xh, (int)MD);
  pack_w_k<<<(D_ * D_) / 256, 256, 0, stream>>>(wq, wqp, D_, D_);
  pack_w_k<<<(D_ * D_) / 256, 256, 0, stream>>>(wk, wkp, D_, D_);
  pack_w_k<<<(D_ * D_) / 256, 256, 0, stream>>>(wv, wvp, D_, D_);
  pack_w_k<<<(D_ * D_) / 256, 256, 0, stream>>>(wo, wop, D_, D_);
  pack_w_k<<<(D_ * FF_) / 256, 256, 0, stream>>>(w1, w1p, D_, FF_);
  pack_w_k<<<(FF_ * D_) / 256, 256, 0, stream>>>(w2, w2p, FF_, D_);

  // 2) QKV projections (f16 outputs for attention WMMA)
  dim3 gblk(128);
  dim3 gproj(D_ / 128, M_ / 64);
  gemm_wmma_k<false, false, true><<<gproj, gblk, 0, stream>>>(xh, wqp, bq, (float*)nullptr, qh, M_, D_, D_);
  gemm_wmma_k<false, false, true><<<gproj, gblk, 0, stream>>>(xh, wkp, bk, (float*)nullptr, kh, M_, D_, D_);
  gemm_wmma_k<false, false, true><<<gproj, gblk, 0, stream>>>(xh, wvp, bv, (float*)nullptr, vh, M_, D_, D_);
  transpose_v_k<<<(int)(MD / 256), 256, 0, stream>>>(vh, vT);

  // 3) fused flash attention
  attention_k<<<dim3(S_ / 64, B_ * H_), gblk, 0, stream>>>(qh, kh, vT, ctx);

  // 4) output projection + residual + LN (also f16 copy for FFN)
  gemm_wmma_k<false, true, false><<<gproj, gblk, 0, stream>>>(ctx, wop, bo, attn_out, (f16*)nullptr, M_, D_, D_);
  add_ln_k<<<M_, 256, 0, stream>>>(x, attn_out, g1, be1, out1, out1h);

  // 5) FFN
  gemm_wmma_k<true, false, true><<<dim3(FF_ / 128, M_ / 64), gblk, 0, stream>>>(
      out1h, w1p, b1, (float*)nullptr, ffh, M_, D_, FF_);
  gemm_wmma_k<false, true, false><<<dim3(D_ / 128, M_ / 64), gblk, 0, stream>>>(
      ffh, w2p, b2, ffn_out, (f16*)nullptr, M_, FF_, D_);

  // 6) final residual + LN -> d_out (fp32)
  add_ln_k<<<M_, 256, 0, stream>>>(out1, ffn_out, g2, be2, out, (f16*)nullptr);
}
